// AbstractEGCN_70909910057016
// MI455X (gfx1250) — matrix operand order repaired and checked by reading, not compiled, not run on hardware
//
#include <hip/hip_runtime.h>
#include <hip/hip_bf16.h>
#include <stdint.h>

#define N_NODES  10000
#define N_EDGES  320000
#define HIDDEN   256
#define KDIM     512          // 2*HIDDEN (concat src,dst)
#define OUTC     10
#define BLK_E    128          // edges per block
#define NTHREADS 256          // 8 waves (wave32)

typedef __bf16 bf16_t;
typedef bf16_t v16bf __attribute__((ext_vector_type(16)));
typedef float  v8f   __attribute__((ext_vector_type(8)));

__device__ __forceinline__ unsigned short f2bf(float f) {
    union { float f; unsigned u; } x; x.f = f;
    unsigned u = x.u;
    u += 0x7fffu + ((u >> 16) & 1u);      // round-to-nearest-even
    return (unsigned short)(u >> 16);
}

// ---------------- pre-pass: x (f32) -> bf16 ----------------
__global__ void cvt_x_kernel(const float* __restrict__ x,
                             unsigned short* __restrict__ xb, int n) {
    int i = blockIdx.x * blockDim.x + threadIdx.x;
    if (i < n) xb[i] = f2bf(x[i]);
}

// ---- pre-pass: pack W1 [512x256] into WMMA-B frag layout (bf16) ----
// dest flat index: ((kb*16 + nb)*32 + lane)*16 + t
//   lane<16 : holds K = kb*32 + (t<8 ? t : t+8)     , N = nb*16 + lane
//   lane>=16: holds K = kb*32 + 8 + (t<8 ? t : t+8) , N = nb*16 + (lane-16)
__global__ void pack_w1_kernel(const float* __restrict__ W1,
                               unsigned short* __restrict__ pw1) {
    int i = blockIdx.x * blockDim.x + threadIdx.x;   // over 512*256
    if (i >= KDIM * HIDDEN) return;
    int k = i / HIDDEN, n = i % HIDDEN;
    int kb = k >> 5, kr = k & 31;
    int nb = n >> 4, nl = n & 15;
    int lane = ((kr & 8) ? 16 : 0) + nl;
    int t    = (kr & 7) | ((kr >> 4) << 3);
    int dst  = (((kb * 16 + nb) * 32 + lane) << 4) + t;
    pw1[dst] = f2bf(W1[i]);
}

// ---- pre-pass: pack W2 [256x10] padded to N=16, WMMA-B frag layout ----
__global__ void pack_w2_kernel(const float* __restrict__ W2,
                               unsigned short* __restrict__ pw2) {
    int i = blockIdx.x * blockDim.x + threadIdx.x;   // 8*32*16 = 4096
    if (i >= 8 * 32 * 16) return;
    int t    = i & 15;
    int lane = (i >> 4) & 31;
    int kb   = i >> 9;
    int n = lane & 15;
    int k = kb * 32 + (lane < 16 ? 0 : 8) + (t < 8 ? t : t + 8);
    pw2[i] = (n < OUTC) ? f2bf(W2[k * OUTC + n]) : (unsigned short)0;
}

// ---------------- main fused kernel ----------------
// LDS layout (dynamic, 102400 B):
//   [0      , 32768)  : sB  double-buffered packed-W1 K-slice (2 x 16 KB)
//                       (reused as sL: 128x16 f32 logits after GEMM1)
//   [32768  , 102400) : sH  h activations, bf16, 128 rows x stride 272
__global__ void __launch_bounds__(NTHREADS)
egcn_main_kernel(const unsigned short* __restrict__ xb,
                 const int* __restrict__ esrc, const int* __restrict__ edst,
                 const unsigned short* __restrict__ pw1,
                 const float* __restrict__ b1,
                 const unsigned short* __restrict__ pw2,
                 const float* __restrict__ b2,
                 float* __restrict__ out) {
    extern __shared__ char smem[];
    unsigned short* sB = (unsigned short*)smem;
    unsigned short* sH = (unsigned short*)(smem + 32768);
    float*          sL = (float*)smem;

    const int tid    = threadIdx.x;
    const int wave   = tid >> 5;
    const int lane   = tid & 31;
    const int lh     = lane & 15;        // row/col-in-tile index
    const int hiHalf = lane >> 4;        // 0 or 1 (K-chunk selector)
    const int eBase  = blockIdx.x * BLK_E;

    // GEMM1 tiling: waves 0..3 -> cols 0..127, waves 4..7 -> cols 128..255
    // each wave: 32 rows (two 16-row M tiles) x 8 N tiles of 16
    const int mw     = (wave & 3) * 32;
    const int nbBase = (wave >> 2) * 8;

    // edge indices for this wave's two M tiles (lane L and L+16 share row L)
    const int e0 = eBase + mw + lh;
    const int e1 = e0 + 16;
    const int s0 = esrc[e0], d0 = edst[e0];
    const int s1 = esrc[e1], d1 = edst[e1];

    v8f acc[2][8];
    {
        const v8f z = {0.f,0.f,0.f,0.f,0.f,0.f,0.f,0.f};
        #pragma unroll
        for (int a = 0; a < 2; ++a)
            #pragma unroll
            for (int b = 0; b < 8; ++b) acc[a][b] = z;
    }

    // ---- stage K-slice kb=0 of packed W1 into LDS buffer 0 ----
    {
        const uint4* g = (const uint4*)pw1;
        uint4* s = (uint4*)sB;
        #pragma unroll
        for (int i = 0; i < 4; ++i) s[tid + i * NTHREADS] = g[tid + i * NTHREADS];
    }
    __syncthreads();

    // ---- K loop: 16 steps of K=32, double-buffered B staging ----
    for (int kb = 0; kb < 16; ++kb) {
        const int cur = kb & 1;

        // prefetch next W1 slice into registers (overlaps WMMA below)
        uint4 nx[4];
        if (kb + 1 < 16) {
            const uint4* g = (const uint4*)(pw1 + (size_t)(kb + 1) * 8192);
            #pragma unroll
            for (int i = 0; i < 4; ++i) nx[i] = g[tid + i * NTHREADS];
        }

        // A fragments: gather from bf16 node features (concat src|dst)
        const int k0   = kb * 32;
        const int kloc = k0 & 255;
        const int row0 = (k0 < 256) ? s0 : d0;
        const int row1 = (k0 < 256) ? s1 : d1;
        const int coff = kloc + (hiHalf ? 8 : 0);   // per-lane K chunk base
        union { uint4 q[2]; v16bf v; } A0, A1;
        {
            const unsigned short* p0 = xb + (size_t)row0 * HIDDEN + coff;
            const unsigned short* p1 = xb + (size_t)row1 * HIDDEN + coff;
            A0.q[0] = *(const uint4*)(p0);
            A0.q[1] = *(const uint4*)(p0 + 16);
            A1.q[0] = *(const uint4*)(p1);
            A1.q[1] = *(const uint4*)(p1 + 16);
        }

        // B fragments from LDS + 16 WMMAs
        const uint4* sBc = (const uint4*)(sB + cur * 8192);
        #pragma unroll
        for (int nb = 0; nb < 8; ++nb) {
            const int nbA = nbBase + nb;
            union { uint4 q[2]; v16bf v; } B;
            B.q[0] = sBc[(nbA * 32 + lane) * 2 + 0];
            B.q[1] = sBc[(nbA * 32 + lane) * 2 + 1];
            acc[0][nb] = __builtin_amdgcn_wmma_f32_16x16x32_bf16(
                false, A0.v, false, B.v, (short)0, acc[0][nb], false, false);
            acc[1][nb] = __builtin_amdgcn_wmma_f32_16x16x32_bf16(
                false, A1.v, false, B.v, (short)0, acc[1][nb], false, false);
        }

        // commit prefetched slice into the other LDS buffer
        if (kb + 1 < 16) {
            uint4* sn = (uint4*)(sB + ((kb + 1) & 1) * 8192);
            #pragma unroll
            for (int i = 0; i < 4; ++i) sn[tid + i * NTHREADS] = nx[i];
        }
        __syncthreads();
    }

    // ---- +b1, ReLU, store h (bf16) to LDS, row stride 272 (16B aligned) ----
    #pragma unroll
    for (int nb = 0; nb < 8; ++nb) {
        const int col = (nbBase + nb) * 16 + lh;   // C/D layout: N = lane%16
        const float bv = b1[col];
        #pragma unroll
        for (int mt = 0; mt < 2; ++mt) {
            #pragma unroll
            for (int r = 0; r < 8; ++r) {
                float v = acc[mt][nb][r] + bv;      // C/D: row = r (+8 hi half)
                v = v > 0.f ? v : 0.f;
                const int row = mw + mt * 16 + r + hiHalf * 8;
                sH[row * 272 + col] = f2bf(v);
            }
        }
    }
    __syncthreads();

    // ---- GEMM2: each wave does 16 rows x 16 cols (10 valid), K=256 ----
    const int m2 = wave * 16;
    v8f cc = {0.f,0.f,0.f,0.f,0.f,0.f,0.f,0.f};
    #pragma unroll
    for (int kb = 0; kb < 8; ++kb) {
        union { uint4 q[2]; v16bf v; } A, B;
        const unsigned short* hp = sH + (m2 + lh) * 272 + kb * 32 + (hiHalf ? 8 : 0);
        A.q[0] = *(const uint4*)(hp);
        A.q[1] = *(const uint4*)(hp + 16);
        const uint4* g = (const uint4*)(pw2 + (size_t)(kb * 32 + lane) * 16);
        B.q[0] = g[0];
        B.q[1] = g[1];
        cc = __builtin_amdgcn_wmma_f32_16x16x32_bf16(
            false, A.v, false, B.v, (short)0, cc, false, false);
    }

    // ---- logits (+b2) -> sL (reuses sB region, free after GEMM1) ----
    const float b2v = (lh < OUTC) ? b2[lh] : 0.f;
    #pragma unroll
    for (int r = 0; r < 8; ++r) {
        const int row = m2 + r + hiHalf * 8;
        sL[row * 16 + lh] = cc[r] + b2v;
    }
    __syncthreads();

    // ---- log_softmax, one thread per edge row ----
    if (tid < BLK_E) {
        float v[OUTC];
        float mx = -3.4e38f;
        #pragma unroll
        for (int j = 0; j < OUTC; ++j) {
            v[j] = sL[tid * 16 + j];
            mx = fmaxf(mx, v[j]);
        }
        float s = 0.f;
        #pragma unroll
        for (int j = 0; j < OUTC; ++j) s += __expf(v[j] - mx);
        const float lse = mx + __logf(s);
        float* o = out + (size_t)(eBase + tid) * OUTC;
        #pragma unroll
        for (int j = 0; j < OUTC; ++j) o[j] = v[j] - lse;
    }
}

// ---------------- host launcher ----------------
extern "C" void kernel_launch(void* const* d_in, const int* in_sizes, int n_in,
                              void* d_out, int out_size, void* d_ws, size_t ws_size,
                              hipStream_t stream) {
    const float* x    = (const float*)d_in[0];
    const int*   eidx = (const int*)d_in[1];      // [2, E]
    const float* W1   = (const float*)d_in[2];    // [512, 256]
    const float* b1   = (const float*)d_in[3];    // [256]
    const float* W2   = (const float*)d_in[4];    // [256, 10]
    const float* b2   = (const float*)d_in[5];    // [10]
    float* out = (float*)d_out;

    // workspace: bf16 node features + packed weights (~5.4 MB)
    unsigned short* xb  = (unsigned short*)d_ws;               // 10000*256
    unsigned short* pw1 = xb  + (size_t)N_NODES * HIDDEN;      // 512*256
    unsigned short* pw2 = pw1 + (size_t)KDIM * HIDDEN;         // 8*32*16

    const int nx = N_NODES * HIDDEN;
    cvt_x_kernel<<<(nx + 255) / 256, 256, 0, stream>>>(x, xb, nx);
    pack_w1_kernel<<<(KDIM * HIDDEN + 255) / 256, 256, 0, stream>>>(W1, pw1);
    pack_w2_kernel<<<(8 * 32 * 16 + 255) / 256, 256, 0, stream>>>(W2, pw2);

    const int* esrc = eidx;
    const int* edst = eidx + N_EDGES;
    const size_t lds_bytes = 32768 + (size_t)BLK_E * 272 * 2;  // 102400
    egcn_main_kernel<<<N_EDGES / BLK_E, NTHREADS, lds_bytes, stream>>>(
        xb, esrc, edst, pw1, b1, pw2, b2, out);
}
